// SRNet_34806414967297
// MI455X (gfx1250) — matrix operand order repaired and checked by reading, not compiled, not run on hardware
//
#include <hip/hip_runtime.h>

typedef __attribute__((ext_vector_type(16))) _Float16 v16h;
typedef __attribute__((ext_vector_type(8)))  float    v8f;

union Frag16 { v16h h; unsigned int u[8]; };

#define HIN  512
#define WIN  512
#define HO   511
#define WO   511
#define TR   8              // tile rows of conv output per block
#define TC   32             // tile cols
#define NPIX (TR*TC)        // 256 pixels per block
#define XR   (TR+5)         // 13 x-rows needed (2x2 conv of 5x5 conv)
#define XC   (TC+5)         // 37
#define RRN  (TR+1)         // 9 rc rows
#define RCN  (TC+1)         // 33 rc cols
#define WOUT 2044

__launch_bounds__(256)
__global__ void srnet_fused(
    const float* __restrict__ x,
    const float* __restrict__ rcw1, const float* __restrict__ rcb1,
    const float* __restrict__ rcw2, const float* __restrict__ rcb2,
    const float* __restrict__ cw1,  const float* __restrict__ cb1,
    const float* __restrict__ cw2,  const float* __restrict__ cb2,
    const float* __restrict__ cw3,  const float* __restrict__ cb3,
    const float* __restrict__ cw4,  const float* __restrict__ cb4,
    const float* __restrict__ cw5,  const float* __restrict__ cb5,
    const float* __restrict__ cw6,  const float* __restrict__ cb6,
    float* __restrict__ out)
{
    __shared__ float s_x[XR*XC];
    __shared__ float s_rc[RRN*RCN];
    __shared__ float s_coeff[25];
    __shared__ float s_tb[25];
    __shared__ float s_w1[64*4];
    __shared__ float s_b1[64];
    alignas(16) __shared__ _Float16 s_W[5][64*64];   // W2..W5, W6(16 rows used)
    __shared__ float s_bias[5][64];
    alignas(16) __shared__ _Float16 s_h[NPIX*64];    // activations [pixel][chan], f16

    const int tid = threadIdx.x;
    const int blk = blockIdx.x;
    const int b  = blk >> 10;           // / (64*16) : image
    const int ti = (blk >> 4) & 63;     // row tile (64 tiles of 8 cover 511)
    const int tj = blk & 15;            // col tile (16 tiles of 32 cover 511)
    const int i0 = ti*TR, j0 = tj*TC;

    // ---- fold RC module into a 5x5 kernel: coeff[k], tbias[k] ----
    if (tid < 25) {
        float cf = 0.f, tb = 0.f;
        for (int c = 0; c < 64; ++c) {
            cf += rcw1[tid*64+c]*rcw2[tid*64+c];
            tb += rcb1[tid*64+c]*rcw2[tid*64+c];
        }
        s_coeff[tid] = cf;
        s_tb[tid]    = tb + rcb2[tid];
    }
    // ---- stage conv1 weights ----
    s_w1[tid] = cw1[tid];                 // 64*1*2*2 = 256 == blockDim
    if (tid < 64) s_b1[tid] = cb1[tid];
    // ---- stage 1x1 weights as f16 [oc][ic] ----
    {
        const float* Wsrc[4] = {cw2, cw3, cw4, cw5};
        const float* Bsrc[4] = {cb2, cb3, cb4, cb5};
        #pragma unroll
        for (int l = 0; l < 4; ++l) {
            for (int idx = tid; idx < 4096; idx += 256)
                s_W[l][idx] = (_Float16)Wsrc[l][idx];
            if (tid < 64) s_bias[l][tid] = Bsrc[l][tid];
        }
        for (int idx = tid; idx < 1024; idx += 256)
            s_W[4][idx] = (_Float16)cw6[idx];
        if (tid < 16) s_bias[4][tid] = cb6[tid];
    }
    // ---- stage x tile (zero-padded) ----
    for (int idx = tid; idx < XR*XC; idx += 256) {
        int rr = idx / XC, cc = idx % XC;
        int gr = i0 - 2 + rr, gc = j0 - 2 + cc;
        float v = 0.f;
        if ((unsigned)gr < (unsigned)HIN && (unsigned)gc < (unsigned)WIN)
            v = x[((size_t)b*HIN + gr)*WIN + gc];
        s_x[idx] = v;
    }
    __syncthreads();

    // ---- rc = mean_k(coeff_k * x_k + tbias_k) : 5x5 window ----
    for (int idx = tid; idx < RRN*RCN; idx += 256) {
        int rr = idx / RCN, cc = idx % RCN;
        float acc = 0.f;
        #pragma unroll
        for (int ki = 0; ki < 5; ++ki)
            #pragma unroll
            for (int kj = 0; kj < 5; ++kj) {
                int k = ki*5 + kj;
                acc += s_coeff[k]*s_x[(rr+ki)*XC + (cc+kj)] + s_tb[k];
            }
        s_rc[idx] = acc * (1.f/25.f);
    }
    __syncthreads();

    // ---- conv1 (2x2 VALID, 1->64) + ReLU -> s_h f16 [p][oc] ----
    {
        int oc = tid & 63;
        int p0 = tid >> 6;
        float w00 = s_w1[oc*4+0], w01 = s_w1[oc*4+1];
        float w10 = s_w1[oc*4+2], w11 = s_w1[oc*4+3];
        float bb  = s_b1[oc];
        for (int p = p0; p < NPIX; p += 4) {
            int r = p >> 5, c = p & 31;
            float v = bb + w00*s_rc[r*RCN+c]     + w01*s_rc[r*RCN+c+1]
                         + w10*s_rc[(r+1)*RCN+c] + w11*s_rc[(r+1)*RCN+c+1];
            s_h[p*64 + oc] = (_Float16)fmaxf(v, 0.f);
        }
    }
    __syncthreads();

    // ---- per-wave WMMA GEMM chain: each wave owns 32 pixels (2 N-tiles) ----
    const int wave = tid >> 5;
    const int lane = tid & 31;
    const int ln   = lane & 15;
    const int hi   = lane >> 4;

    // conv2..conv5: 64->64, ReLU applied when storing back
    for (int l = 0; l < 4; ++l) {
        Frag16 Bf[2][2];
        #pragma unroll
        for (int nt = 0; nt < 2; ++nt) {
            int p = wave*32 + nt*16 + ln;
            #pragma unroll
            for (int kk = 0; kk < 2; ++kk)
                #pragma unroll
                for (int v = 0; v < 8; ++v) {
                    int K = kk*32 + hi*16 + 2*v;   // B 32x16 f16 layout
                    Bf[nt][kk].u[v] = *(const unsigned int*)&s_h[p*64 + K];
                }
        }
        #pragma unroll
        for (int mt = 0; mt < 4; ++mt) {
            Frag16 Af[2];
            #pragma unroll
            for (int kk = 0; kk < 2; ++kk)
                #pragma unroll
                for (int v = 0; v < 8; ++v) {       // A 16x32 f16 layout
                    int K = kk*32 + hi*16 + 2*v;
                    Af[kk].u[v] = *(const unsigned int*)&s_W[l][(mt*16+ln)*64 + K];
                }
            #pragma unroll
            for (int nt = 0; nt < 2; ++nt) {
                v8f acc;
                #pragma unroll
                for (int v = 0; v < 8; ++v)
                    acc[v] = s_bias[l][mt*16 + v + 8*hi];
                acc = __builtin_amdgcn_wmma_f32_16x16x32_f16(
                          false, Af[0].h, false, Bf[nt][0].h,
                          (short)0, acc, false, false);
                acc = __builtin_amdgcn_wmma_f32_16x16x32_f16(
                          false, Af[1].h, false, Bf[nt][1].h,
                          (short)0, acc, false, false);
                union { uint4 q; _Float16 hh[8]; } pk;
                #pragma unroll
                for (int v = 0; v < 8; ++v)
                    pk.hh[v] = (_Float16)fmaxf(acc[v], 0.f);   // ReLU for next layer
                int p = wave*32 + nt*16 + ln;
                *(uint4*)&s_h[p*64 + mt*16 + 8*hi] = pk.q;     // 8 f16, 16B aligned
            }
        }
    }

    // conv6: 64->16, no ReLU; fuse pixel-shuffle store
    {
        Frag16 Af[2];
        #pragma unroll
        for (int kk = 0; kk < 2; ++kk)
            #pragma unroll
            for (int v = 0; v < 8; ++v) {
                int K = kk*32 + hi*16 + 2*v;
                Af[kk].u[v] = *(const unsigned int*)&s_W[4][ln*64 + K];
            }
        #pragma unroll
        for (int nt = 0; nt < 2; ++nt) {
            int p = wave*32 + nt*16 + ln;
            Frag16 B0, B1;
            #pragma unroll
            for (int v = 0; v < 8; ++v) {
                B0.u[v] = *(const unsigned int*)&s_h[p*64 +      hi*16 + 2*v];
                B1.u[v] = *(const unsigned int*)&s_h[p*64 + 32 + hi*16 + 2*v];
            }
            v8f acc;
            #pragma unroll
            for (int v = 0; v < 8; ++v)
                acc[v] = s_bias[4][v + 8*hi];
            acc = __builtin_amdgcn_wmma_f32_16x16x32_f16(
                      false, Af[0].h, false, B0.h, (short)0, acc, false, false);
            acc = __builtin_amdgcn_wmma_f32_16x16x32_f16(
                      false, Af[1].h, false, B1.h, (short)0, acc, false, false);
            // pixel shuffle r=4: oc = v + 8*hi -> (ry,rx) = (oc>>2, oc&3)
            int r = p >> 5, c = p & 31;
            int i = i0 + r, j = j0 + c;
            if (i < HO && j < WO) {
                float* o = out + ((size_t)b*WOUT + (4*i + 2*hi))*WOUT + 4*j;
                float4 lo; lo.x = acc[0]; lo.y = acc[1]; lo.z = acc[2]; lo.w = acc[3];
                float4 up; up.x = acc[4]; up.y = acc[5]; up.z = acc[6]; up.w = acc[7];
                *(float4*)o          = lo;   // row 4i+2*hi
                *(float4*)(o + WOUT) = up;   // row 4i+2*hi+1
            }
        }
    }
}

extern "C" void kernel_launch(void* const* d_in, const int* in_sizes, int n_in,
                              void* d_out, int out_size, void* d_ws, size_t ws_size,
                              hipStream_t stream) {
    (void)in_sizes; (void)n_in; (void)out_size; (void)d_ws; (void)ws_size;
    const float* x    = (const float*)d_in[0];
    const float* rcw1 = (const float*)d_in[1];
    const float* rcb1 = (const float*)d_in[2];
    const float* rcw2 = (const float*)d_in[3];
    const float* rcb2 = (const float*)d_in[4];
    const float* cw1  = (const float*)d_in[5];
    const float* cb1  = (const float*)d_in[6];
    const float* cw2  = (const float*)d_in[7];
    const float* cb2  = (const float*)d_in[8];
    const float* cw3  = (const float*)d_in[9];
    const float* cb3  = (const float*)d_in[10];
    const float* cw4  = (const float*)d_in[11];
    const float* cb4  = (const float*)d_in[12];
    const float* cw5  = (const float*)d_in[13];
    const float* cb5  = (const float*)d_in[14];
    const float* cw6  = (const float*)d_in[15];
    const float* cb6  = (const float*)d_in[16];
    float* out = (float*)d_out;

    // 8 images x 64 row-tiles (8 rows) x 16 col-tiles (32 cols)
    dim3 grid(8 * 64 * 16);
    dim3 block(256);
    srnet_fused<<<grid, block, 0, stream>>>(x, rcw1, rcb1, rcw2, rcb2,
                                            cw1, cb1, cw2, cb2, cw3, cb3,
                                            cw4, cb4, cw5, cb5, cw6, cb6, out);
}
